// GINEEncoderBlock_1975684956226
// MI455X (gfx1250) — matrix operand order repaired and checked by reading, not compiled, not run on hardware
//
#include <hip/hip_runtime.h>
#include <hip/hip_bf16.h>

// GINE encoder block for MI455X (gfx1250, wave32).
// bf16 WMMA (f32 accumulate) for all GEMMs; bf16 storage for the edge tensor
// (the only >L2 stream); fp32 everywhere accumulation matters.

typedef __bf16 bf16_t;
typedef __attribute__((ext_vector_type(16))) __bf16 v16bf;
typedef __attribute__((ext_vector_type(8)))  __bf16 v8bf;
typedef __attribute__((ext_vector_type(4)))  __bf16 v4bf;
typedef __attribute__((ext_vector_type(8)))  float  v8f;

#define DD 256   // hidden dim (fixed by the reference)

// ---------------------------------------------------------------- utilities

__global__ void k_zero(float* __restrict__ p, int n) {
  int i = blockIdx.x * blockDim.x + threadIdx.x;
  if (i < n) p[i] = 0.f;
}

// W[l][k][n] (fp32) -> Wt[l][n][k] (bf16) for both weight stacks.
__global__ void k_prep_weights(const float* __restrict__ Wc, const float* __restrict__ We,
                               bf16_t* __restrict__ WtC, bf16_t* __restrict__ WtE, int total) {
  int i = blockIdx.x * blockDim.x + threadIdx.x;
  if (i >= total) return;
  int l = i >> 16;            // D*D == 65536
  int r = i & 0xFFFF;
  int k = r >> 8;
  int n = r & 0xFF;
  int o = (l << 16) + (n << 8) + k;
  WtC[o] = (bf16_t)Wc[i];
  WtE[o] = (bf16_t)We[i];
}

__global__ void k_convert_edges(const float* __restrict__ ef, bf16_t* __restrict__ eb, int total4) {
  int i = blockIdx.x * blockDim.x + threadIdx.x;
  if (i >= total4) return;
  float4 v = ((const float4*)ef)[i];
  v4bf o;
  o[0] = (bf16_t)v.x; o[1] = (bf16_t)v.y; o[2] = (bf16_t)v.z; o[3] = (bf16_t)v.w;
  ((v4bf*)eb)[i] = o;
}

// xin_bf16 = bf16(x + aggr)
__global__ void k_fuse_input(const float* __restrict__ x, const float* __restrict__ aggr,
                             bf16_t* __restrict__ xin, int total4) {
  int i = blockIdx.x * blockDim.x + threadIdx.x;
  if (i >= total4) return;
  float4 a = ((const float4*)x)[i];
  float4 b = ((const float4*)aggr)[i];
  v4bf o;
  o[0] = (bf16_t)(a.x + b.x); o[1] = (bf16_t)(a.y + b.y);
  o[2] = (bf16_t)(a.z + b.z); o[3] = (bf16_t)(a.w + b.w);
  ((v4bf*)xin)[i] = o;
}

// ------------------------------------------------------- message + scatter
// One wave per edge; stride-32 element mapping keeps loads and the
// global_atomic_add_f32 packets fully coalesced. aggr (10 MB) is L2-resident.
__global__ void k_message(const float* __restrict__ x, const bf16_t* __restrict__ e,
                          const int* __restrict__ src, const int* __restrict__ dst,
                          float* __restrict__ aggr, int E) {
  int eid = blockIdx.x * (blockDim.x >> 5) + (threadIdx.x >> 5);
  if (eid >= E) return;                                  // wave-uniform exit
  int lane = threadIdx.x & 31;
  const float*  xr = x + (size_t)src[eid] * DD;
  const bf16_t* er = e + (size_t)eid * DD;
  float*        ar = aggr + (size_t)dst[eid] * DD;
#pragma unroll
  for (int j = 0; j < 8; j++) {
    int c = lane + 32 * j;
    float m = xr[c] + (float)er[c];
    m = fmaxf(m, 0.f);
    __hip_atomic_fetch_add(&ar[c], m, __ATOMIC_RELAXED, __HIP_MEMORY_SCOPE_AGENT);
  }
}

// -------------------------------------------------------------------- GEMM
// One 16-row x 256-col output strip per wave (wave32), 8 waves/block.
// A panel (16 rows x K=256, bf16) held entirely in 64 VGPRs and reused across
// all 16 column tiles; B streamed from the L2-hot 128 KB transposed weights.
// Per column tile: all 8 B fragments are preloaded (one 16-load clause) so the
// 8 WMMAs issue back-to-back instead of load->wait->wmma serialization.
//
// Fragment layouts (CDNA5 ISA 7.12.2, "row-major -> B128 load" rule):
//   A 16x32: lane l=lane&15,h=lane>>4 : row m0+l, two b128 loads at K=kb*32+h*8
//            and K=kb*32+16+h*8.
//   B 32x16 (from Wt[n][k]): col n0+l, b128 loads at K=kb*32+h*8 and K=kb*32+16+h*8.
//   C/D v8f: VGPR r holds (row m0+h*8+r, col n0+l).

__device__ __forceinline__ v16bf ld_frag(const bf16_t* lo, const bf16_t* hi) {
  v8bf a = *(const v8bf*)lo;
  v8bf b = *(const v8bf*)hi;
  return __builtin_shufflevector(a, b, 0, 1, 2, 3, 4, 5, 6, 7,
                                 8, 9, 10, 11, 12, 13, 14, 15);
}

// MODE 0: node hidden  -> f32_out = relu(.), fused BN column sum/sumsq atomics
// MODE 1: edge update  -> bf16_out = relu(.)   (in place over A is safe: each
//                         wave's A panel is fully in registers before stores)
// MODE 2: final conv   -> f32_out = relu(.) + addsrc (residual), to d_out
template <int MODE>
__global__ void k_gemm(const bf16_t* A, const bf16_t* __restrict__ Wt,
                       const float* __restrict__ bias, int ntiles,
                       float* __restrict__ f32_out, bf16_t* bf16_out,
                       float* __restrict__ colsum, float* __restrict__ colsumsq,
                       const float* __restrict__ addsrc) {
  int tile = blockIdx.x * (blockDim.x >> 5) + (threadIdx.x >> 5);
  if (tile >= ntiles) return;                            // wave-uniform: EXEC stays all-1s
  int lane = threadIdx.x & 31;
  int l = lane & 15, half = lane >> 4;
  int m0 = tile << 4;

  const bf16_t* arow = A + (size_t)(m0 + l) * DD;
  v16bf a[8];
#pragma unroll
  for (int kb = 0; kb < 8; kb++)
    a[kb] = ld_frag(arow + kb * 32 + half * 8, arow + kb * 32 + 16 + half * 8);

  const bf16_t* bbase = Wt + (size_t)l * DD + half * 8;

#pragma unroll 1
  for (int n0 = 0; n0 < DD; n0 += 16) {
    const bf16_t* bcol = bbase + (size_t)n0 * DD;
    int col = n0 + l;
    float bc = bias[col];

    // Preload the whole K=256 strip of B fragments (one load clause),
    // then run the 8 WMMAs back-to-back.
    v16bf b[8];
#pragma unroll
    for (int kb = 0; kb < 8; kb++)
      b[kb] = ld_frag(bcol + kb * 32, bcol + kb * 32 + 16);

    v8f acc = {0.f, 0.f, 0.f, 0.f, 0.f, 0.f, 0.f, 0.f};
#pragma unroll
    for (int kb = 0; kb < 8; kb++)
      acc = __builtin_amdgcn_wmma_f32_16x16x32_bf16(false, a[kb], false, b[kb],
                                                    (short)0, acc, false, false);

    float s = 0.f, sq = 0.f;
    size_t base = (size_t)(m0 + half * 8) * DD + col;
#pragma unroll
    for (int r = 0; r < 8; r++) {
      float v = acc[r] + bc;
      v = fmaxf(v, 0.f);
      size_t o = base + (size_t)r * DD;
      if (MODE == 0)      { f32_out[o] = v; s += v; sq += v * v; }
      else if (MODE == 1) { bf16_out[o] = (bf16_t)v; }
      else                { f32_out[o] = v + addsrc[o]; }
    }
    if (MODE == 0) {
      s  += __shfl_xor(s, 16);
      sq += __shfl_xor(sq, 16);
      if (half == 0) {
        __hip_atomic_fetch_add(&colsum[col],   s,  __ATOMIC_RELAXED, __HIP_MEMORY_SCOPE_AGENT);
        __hip_atomic_fetch_add(&colsumsq[col], sq, __ATOMIC_RELAXED, __HIP_MEMORY_SCOPE_AGENT);
      }
    }
  }
}

// BatchNorm (training-mode stats), biased variance like jnp.var.
__global__ void k_bn(const float* __restrict__ h, const float* __restrict__ colsum,
                     const float* __restrict__ colsumsq, const float* __restrict__ gamma,
                     const float* __restrict__ beta, float* __restrict__ xout,
                     int n, float invN) {
  int i = blockIdx.x * blockDim.x + threadIdx.x;
  if (i >= n) return;
  int c = i & (DD - 1);
  float mu  = colsum[c] * invN;
  float var = colsumsq[c] * invN - mu * mu;
  float rs  = rsqrtf(var + 1e-5f);
  xout[i] = (h[i] - mu) * rs * gamma[c] + beta[c];
}

// ------------------------------------------------------------------- launch

extern "C" void kernel_launch(void* const* d_in, const int* in_sizes, int n_in,
                              void* d_out, int out_size, void* d_ws, size_t ws_size,
                              hipStream_t stream) {
  const float* node_feat = (const float*)d_in[0];
  const float* edge_feat = (const float*)d_in[1];
  const int*   src       = (const int*)d_in[2];
  const int*   dst       = (const int*)d_in[3];
  const float* W_conv    = (const float*)d_in[4];
  const float* b_conv    = (const float*)d_in[5];
  const float* W_edge    = (const float*)d_in[6];
  const float* b_edge    = (const float*)d_in[7];
  const float* gamma     = (const float*)d_in[8];
  const float* beta      = (const float*)d_in[9];

  const int N    = in_sizes[0] / DD;
  const int E    = in_sizes[2];
  const int Wtot = in_sizes[4];          // L*D*D
  const int L    = Wtot / (DD * DD);

  // workspace carve-up (~190 MB total)
  char* ws = (char*)d_ws;
  size_t off = 0;
  auto take = [&](size_t bytes) -> char* {
    char* p = ws + off;
    off = (off + bytes + 255) & ~(size_t)255;
    return p;
  };
  bf16_t* e_bf  = (bf16_t*)take((size_t)E * DD * 2);                 // edge feats, bf16
  bf16_t* WtC   = (bf16_t*)take((size_t)Wtot * 2);                   // conv weights, transposed bf16
  bf16_t* WtE   = (bf16_t*)take((size_t)Wtot * 2);                   // edge weights, transposed bf16
  float*  aggr  = (float*)take((size_t)N * DD * 4 + 2 * DD * 4);     // aggr + colsum + colsumsq (contiguous)
  float*  colsum = aggr + (size_t)N * DD;
  float*  colsq  = colsum + DD;
  float*  x_cur = (float*)take((size_t)N * DD * 4);
  float*  h_buf = (float*)take((size_t)N * DD * 4);
  bf16_t* xin   = (bf16_t*)take((size_t)N * DD * 2);

  const int TB = 256;                    // 8 wave32 waves per block
  const int tot4e = E * DD / 4;
  const int tot4n = N * DD / 4;
  const int zn = N * DD + 2 * DD;
  const int ntilesN = (N + 15) / 16;
  const int ntilesE = (E + 15) / 16;
  const int mblocks = (E + 7) / 8;
  const float invN = 1.0f / (float)N;

  // prep (re-done every call: deterministic, no cached state)
  k_prep_weights<<<(Wtot + TB - 1) / TB, TB, 0, stream>>>(W_conv, W_edge, WtC, WtE, Wtot);
  k_convert_edges<<<(tot4e + TB - 1) / TB, TB, 0, stream>>>(edge_feat, e_bf, tot4e);

  const float* x = node_feat;
  for (int i = 0; i < L; i++) {
    k_zero<<<(zn + TB - 1) / TB, TB, 0, stream>>>(aggr, zn);
    k_message<<<mblocks, TB, 0, stream>>>(x, e_bf, src, dst, aggr, E);
    k_fuse_input<<<(tot4n + TB - 1) / TB, TB, 0, stream>>>(x, aggr, xin, tot4n);
    k_gemm<0><<<(ntilesN + 7) / 8, TB, 0, stream>>>(xin, WtC + (size_t)i * DD * DD,
                                                    b_conv + i * DD, ntilesN,
                                                    h_buf, nullptr, colsum, colsq, nullptr);
    k_bn<<<(N * DD + TB - 1) / TB, TB, 0, stream>>>(h_buf, colsum, colsq,
                                                    gamma + i * DD, beta + i * DD,
                                                    x_cur, N * DD, invN);
    k_gemm<1><<<(ntilesE + 7) / 8, TB, 0, stream>>>(e_bf, WtE + (size_t)i * DD * DD,
                                                    b_edge + i * DD, ntilesE,
                                                    nullptr, e_bf, nullptr, nullptr, nullptr);
    x = x_cur;
  }

  // final conv (shares layer L-1 weights) + residual with the original input
  k_zero<<<(zn + TB - 1) / TB, TB, 0, stream>>>(aggr, zn);
  k_message<<<mblocks, TB, 0, stream>>>(x, e_bf, src, dst, aggr, E);
  k_fuse_input<<<(tot4n + TB - 1) / TB, TB, 0, stream>>>(x, aggr, xin, tot4n);
  k_gemm<2><<<(ntilesN + 7) / 8, TB, 0, stream>>>(xin, WtC + (size_t)(L - 1) * DD * DD,
                                                  b_conv + (L - 1) * DD, ntilesN,
                                                  (float*)d_out, nullptr, nullptr, nullptr,
                                                  node_feat);
}
